// EdgeDecoder_50345606644319
// MI455X (gfx1250) — matrix operand order repaired and checked by reading, not compiled, not run on hardware
//
#include <hip/hip_runtime.h>

typedef _Float16 half_t;
typedef __attribute__((ext_vector_type(4)))  half_t h4;
typedef __attribute__((ext_vector_type(8)))  half_t h8;
typedef __attribute__((ext_vector_type(16))) half_t h16;
typedef __attribute__((ext_vector_type(8)))  float  f8;
typedef __attribute__((ext_vector_type(4)))  float  f4v;

#define WPAD   136   // halfs per W1^T row: 128 + 8 pad -> 272B stride
#define RPAD   132   // floats per A-tile row: 128 + 4 pad -> 528B stride
#define NWAVES 8

// One async 16B global->LDS transfer; imm offset applies to BOTH addresses.
#define ASYNC_B128(LDSA, GA, OFF)                                          \
  asm volatile("global_load_async_to_lds_b128 %0, %1, off offset:" #OFF    \
               :: "v"(LDSA), "v"(GA) : "memory")

__global__ __launch_bounds__(256) void edge_decoder_kernel(
    const float* __restrict__ z,
    const long long* __restrict__ eidx,   // [2, nE] int64
    const float* __restrict__ W1,         // [128, 64] row-major
    const float* __restrict__ b1,         // [64]
    const float* __restrict__ W2,         // [64, 1]
    const float* __restrict__ b2,         // [1]
    float* __restrict__ out,              // [nE]
    int nE, int nTiles)
{
  __shared__ half_t sW1T[64 * WPAD];                 // W1^T, f16: [n][k]
  __shared__ float  sAf[NWAVES][2][16 * RPAD];       // double-buffered f32 A tiles

  const int tid  = threadIdx.x;
  const int wave = tid >> 5;
  const int lane = tid & 31;
  const int nl   = lane & 15;   // edge-in-tile / column-in-tile
  const int hk   = lane >> 4;   // half-wave selector

  // ---- stage W1^T into LDS as f16 (once per block) ----
  for (int i = tid; i < 128 * 64; i += 256) {
    int k = i >> 6;
    int n = i & 63;
    sW1T[n * WPAD + k] = (half_t)W1[i];
  }
  __syncthreads();

  // ---- all 16 B fragments resident in VGPRs ----
  // B k order used here (contiguous, same as A): lane = n + 16*hk holds K = kb*32 + hk*16 + (0..15)
  h16 bfrag[16];
#pragma unroll
  for (int kb = 0; kb < 4; ++kb) {
#pragma unroll
    for (int nt = 0; nt < 4; ++nt) {
      const half_t* p = &sW1T[(nt * 16 + nl) * WPAD + kb * 32 + hk * 16];
      h8 lo = *(const h8*)p;
      h8 hi = *(const h8*)(p + 8);
      bfrag[kb * 4 + nt] = __builtin_shufflevector(
          lo, hi, 0, 1, 2, 3, 4, 5, 6, 7, 8, 9, 10, 11, 12, 13, 14, 15);
    }
  }

  // loop-invariant epilogue params (per-lane column n = nt*16 + nl)
  float b1v[4], w2v[4];
#pragma unroll
  for (int nt = 0; nt < 4; ++nt) {
    b1v[nt] = b1[nt * 16 + nl];
    w2v[nt] = W2[nt * 16 + nl];
  }
  const float b2s = b2[0];

  const int gw      = blockIdx.x * NWAVES + wave;
  const int gstride = gridDim.x * NWAVES;

  // Non-temporal streaming read of this lane's node index for tile tt (clamped).
  auto load_node = [&](long long tt) -> long long {
    long long gid = tt * 16 + nl;
    if (gid >= nE) gid = (long long)nE - 1;          // clamp (safe overrun prefetch)
    return __builtin_nontemporal_load(&eidx[(long long)hk * nE + gid]);
  };

  // Issue the 16-edge gather into LDS buffer `buf` (async, no VGPR data traffic).
  auto issue_gather = [&](long long node, float* buf) {
    unsigned long long ga =
        (unsigned long long)(const void*)(z + node * 64);
    unsigned la =
        (unsigned)(unsigned long long)(const void*)(buf + nl * RPAD + hk * 64);
    ASYNC_B128(la, ga, 0);    ASYNC_B128(la, ga, 16);
    ASYNC_B128(la, ga, 32);   ASYNC_B128(la, ga, 48);
    ASYNC_B128(la, ga, 64);   ASYNC_B128(la, ga, 80);
    ASYNC_B128(la, ga, 96);   ASYNC_B128(la, ga, 112);
    ASYNC_B128(la, ga, 128);  ASYNC_B128(la, ga, 144);
    ASYNC_B128(la, ga, 160);  ASYNC_B128(la, ga, 176);
    ASYNC_B128(la, ga, 192);  ASYNC_B128(la, ga, 208);
    ASYNC_B128(la, ga, 224);  ASYNC_B128(la, ga, 240);
  };

  if (gw < nTiles) {
    issue_gather(load_node(gw), &sAf[wave][0][0]);     // prologue fill of buffer 0
    long long node_next = load_node((long long)gw + gstride);  // in flight
    int buf = 0;
    for (int t = gw; t < nTiles; t += gstride) {
      // ---- prefetch next tile into the other buffer (node index already here) ----
      asm volatile("s_wait_dscnt 0x0" ::: "memory");   // close WAR vs old ds_loads
      issue_gather(node_next, &sAf[wave][buf ^ 1][0]);
      node_next = load_node((long long)t + 2LL * gstride);  // hide behind compute

      // current buffer's 16 async b128s are the oldest outstanding -> drain to <=16
      asm volatile("s_wait_asynccnt 0x10" ::: "memory");

      // ---- [16x128] @ [128x64] via 16x v_wmma_f32_16x16x32_f16 ----
      const float* bf = &sAf[wave][buf][0];
      f8 acc0 = {}, acc1 = {}, acc2 = {}, acc3 = {};
#pragma unroll
      for (int kb = 0; kb < 4; ++kb) {
        // A k order used here (contiguous, same as B): lane = m + 16*hk holds K = kb*32+hk*16+(0..15)
        const float* ap = bf + nl * RPAD + kb * 32 + hk * 16;
        f4v x0 = *(const f4v*)(ap);
        f4v x1 = *(const f4v*)(ap + 4);
        f4v x2 = *(const f4v*)(ap + 8);
        f4v x3 = *(const f4v*)(ap + 12);
        h4 c0 = __builtin_convertvector(x0, h4);
        h4 c1 = __builtin_convertvector(x1, h4);
        h4 c2 = __builtin_convertvector(x2, h4);
        h4 c3 = __builtin_convertvector(x3, h4);
        h8 lo = __builtin_shufflevector(c0, c1, 0, 1, 2, 3, 4, 5, 6, 7);
        h8 hi = __builtin_shufflevector(c2, c3, 0, 1, 2, 3, 4, 5, 6, 7);
        h16 a = __builtin_shufflevector(
            lo, hi, 0, 1, 2, 3, 4, 5, 6, 7, 8, 9, 10, 11, 12, 13, 14, 15);
        acc0 = __builtin_amdgcn_wmma_f32_16x16x32_f16(false, a, false, bfrag[kb * 4 + 0],
                                                      (short)0, acc0, false, false);
        acc1 = __builtin_amdgcn_wmma_f32_16x16x32_f16(false, a, false, bfrag[kb * 4 + 1],
                                                      (short)0, acc1, false, false);
        acc2 = __builtin_amdgcn_wmma_f32_16x16x32_f16(false, a, false, bfrag[kb * 4 + 2],
                                                      (short)0, acc2, false, false);
        acc3 = __builtin_amdgcn_wmma_f32_16x16x32_f16(false, a, false, bfrag[kb * 4 + 3],
                                                      (short)0, acc3, false, false);
      }

      // ---- bias + ReLU + [64]->[1] GEMV + half-wave reduction ----
      // C layout: VGPR r holds row m = r + 8*hk; lane's column n = nt*16 + nl
      float part[8];
#pragma unroll
      for (int r = 0; r < 8; ++r) {
        float h0 = fmaxf(acc0[r] + b1v[0], 0.0f);
        float h1 = fmaxf(acc1[r] + b1v[1], 0.0f);
        float h2 = fmaxf(acc2[r] + b1v[2], 0.0f);
        float h3 = fmaxf(acc3[r] + b1v[3], 0.0f);
        float p = h0 * w2v[0] + h1 * w2v[1] + h2 * w2v[2] + h3 * w2v[3];
        p += __shfl_xor(p, 1, 16);
        p += __shfl_xor(p, 2, 16);
        p += __shfl_xor(p, 4, 16);
        p += __shfl_xor(p, 8, 16);
        part[r] = p + b2s;
      }
      if (nl == 0) {                      // lane 0 -> rows 0..7, lane 16 -> rows 8..15
        int base = t * 16 + hk * 8;
#pragma unroll
        for (int r = 0; r < 8; ++r)
          if (base + r < nE)
            __builtin_nontemporal_store(part[r], &out[base + r]);  // write-once stream
      }
      buf ^= 1;
    }
  }
  // s_endpgm performs an implicit wait-idle, draining the trailing prefetch.
}

extern "C" void kernel_launch(void* const* d_in, const int* in_sizes, int n_in,
                              void* d_out, int out_size, void* d_ws, size_t ws_size,
                              hipStream_t stream) {
  const float*     z  = (const float*)d_in[0];
  const long long* ei = (const long long*)d_in[1];   // int64 per reference
  const float*     W1 = (const float*)d_in[2];
  const float*     b1 = (const float*)d_in[3];
  const float*     W2 = (const float*)d_in[4];
  const float*     b2 = (const float*)d_in[5];
  float* out = (float*)d_out;

  int nE     = in_sizes[1] / 2;            // edge_index is [2, nE]
  int nTiles = (nE + 15) / 16;
  int blocks = (nTiles + NWAVES - 1) / NWAVES;
  if (blocks > 4096) blocks = 4096;        // grid-stride; amortize W1 staging

  edge_decoder_kernel<<<blocks, 256, 0, stream>>>(z, ei, W1, b1, W2, b2, out, nE, nTiles);
}
